// LinearAttention_19421842112583
// MI455X (gfx1250) — matrix operand order repaired and checked by reading, compiled, not run on hardware
//
#include <hip/hip_runtime.h>
#include <hip/hip_bf16.h>

typedef __bf16 bf16_t;
typedef __attribute__((ext_vector_type(16))) __bf16 v16bf;
typedef __attribute__((ext_vector_type(8)))  __bf16 v8bf;
typedef __attribute__((ext_vector_type(4)))  __bf16 v4bf;
typedef __attribute__((ext_vector_type(8)))  float  v8f;
typedef __attribute__((ext_vector_type(4)))  float  v4f;
typedef __attribute__((ext_vector_type(4)))  int    v4i;

#define LDS_STRIDE 56   // bf16 elements per staged row: 112B, 16B-aligned, bank-conflict-free

// ---------------------------------------------------------------------------
// WMMA helper: D = A(16x32 bf16) * B(32x16 bf16) + C(16x16 f32)
// ---------------------------------------------------------------------------
__device__ __forceinline__ v8f wmma_bf16(v16bf a, v16bf b, v8f c) {
    return __builtin_amdgcn_wmma_f32_16x16x32_bf16(
        false, a, false, b, (short)0, c, false, false);
}

// ---------------------------------------------------------------------------
// Async global -> LDS copy (CDNA5 GLOBAL_LOAD_ASYNC_TO_LDS_B128, ASYNCcnt)
// Builtin signature: (int4 AS1*, int4 AS3*, imm offset, imm cpol)
// ---------------------------------------------------------------------------
__device__ __forceinline__ void async_b128(const bf16_t* g, bf16_t* l) {
#if __has_builtin(__builtin_amdgcn_global_load_async_to_lds_b128)
    __builtin_amdgcn_global_load_async_to_lds_b128(
        (__attribute__((address_space(1))) v4i*)(v4i*)g,
        (__attribute__((address_space(3))) v4i*)(v4i*)l, 0, 0);
#else
    *(v8bf*)l = *(const v8bf*)g;   // synchronous fallback
#endif
}

template <int N>
__device__ __forceinline__ void wait_async() {
#if __has_builtin(__builtin_amdgcn_s_wait_asynccnt)
    __builtin_amdgcn_s_wait_asynccnt(N);
#else
    asm volatile("s_wait_asynccnt %0" :: "n"(N) : "memory");
#endif
}

// ---------------------------------------------------------------------------
// Fragment loaders (ISA 7.12.2 layouts)
// A 16x32: lane L (m=L&15, h=L>>4): e<8 -> k=8h+e ; e>=8 -> k=16+8h+(e-8)
// B 32x16 from row-major [N,K] source: lane L: n=n0+(L&15), same k order as A
// ---------------------------------------------------------------------------
__device__ __forceinline__ v16bf frag_from2(v8bf lo, v8bf hi) {
    v16bf f;
#pragma unroll
    for (int i = 0; i < 8; ++i) { f[i] = lo[i]; f[i + 8] = hi[i]; }
    return f;
}

__device__ __forceinline__ v16bf loadA_g(const bf16_t* __restrict__ p,
                                         int row0, int k0, int ld) {
    int lane = threadIdx.x & 31;
    const bf16_t* r = p + (long)(row0 + (lane & 15)) * ld + k0 + 8 * (lane >> 4);
    return frag_from2(*(const v8bf*)r, *(const v8bf*)(r + 16));
}

__device__ __forceinline__ v16bf loadB_g(const bf16_t* __restrict__ p,
                                         int n0, int k0, int ld) {
    int lane = threadIdx.x & 31;
    const bf16_t* r = p + (long)(n0 + (lane & 15)) * ld + k0 + 8 * (lane >> 4);
    return frag_from2(*(const v8bf*)r, *(const v8bf*)(r + 16));
}

__device__ __forceinline__ v16bf loadA_lds(const bf16_t* s, int row0) {
    int lane = threadIdx.x & 31;
    const bf16_t* r = s + (row0 + (lane & 15)) * LDS_STRIDE + 8 * (lane >> 4);
    return frag_from2(*(const v8bf*)r, *(const v8bf*)(r + 16));
}

__device__ __forceinline__ v16bf loadB_lds(const bf16_t* s, int n0) {
    int lane = threadIdx.x & 31;
    const bf16_t* r = s + (n0 + (lane & 15)) * LDS_STRIDE + 8 * (lane >> 4);
    return frag_from2(*(const v8bf*)r, *(const v8bf*)(r + 16));
}

// Stage one 128x32 A tile + 128x32 B tile into LDS (16KB), 4 async ops/wave.
__device__ __forceinline__ void stage_copy(bf16_t* sA, bf16_t* sB,
                                           const bf16_t* gA, const bf16_t* gB,
                                           int k0) {
    int t = threadIdx.x;
#pragma unroll
    for (int j = 0; j < 2; ++j) {
        int c = j * 256 + t;       // 16B chunk id, 0..511
        int row = c >> 2;          // 0..127
        int col = c & 3;           // 0..3
        async_b128(gA + (long)row * 1024 + k0 + col * 8, sA + row * LDS_STRIDE + col * 8);
        async_b128(gB + (long)row * 1024 + k0 + col * 8, sB + row * LDS_STRIDE + col * 8);
    }
}

__device__ __forceinline__ float elu1(float v) {
    return v > 0.f ? v + 1.f : __expf(v);
}

// ---------------------------------------------------------------------------
// Kernel 0: f32 -> bf16 conversion, 4 elements/thread
// ---------------------------------------------------------------------------
__global__ void cvt_bf16x4_kernel(const float* __restrict__ in,
                                  bf16_t* __restrict__ out, long n4) {
    long i = (long)blockIdx.x * blockDim.x + threadIdx.x;
    if (i < n4) {
        v4f f = ((const v4f*)in)[i];
        v4bf o;
#pragma unroll
        for (int j = 0; j < 4; ++j) o[j] = (bf16_t)f[j];
        ((v4bf*)out)[i] = o;
    }
}

// ---------------------------------------------------------------------------
// Kernel 1: qkv = xb @ Wqkv^T  (M=32768, N=3072, K=1024), LDS double-buffered
// Epilogue: Q -> elu+1 row-major; K -> elu+1 transposed KT; V transposed VT
// ---------------------------------------------------------------------------
__global__ __launch_bounds__(256) void qkv_gemm_kernel(
    const bf16_t* __restrict__ xb, const bf16_t* __restrict__ Wb,
    bf16_t* __restrict__ Qb, bf16_t* __restrict__ KT, bf16_t* __restrict__ VT) {
    __shared__ bf16_t sA[2][128 * LDS_STRIDE];
    __shared__ bf16_t sB[2][128 * LDS_STRIDE];

    int lane = threadIdx.x & 31;
    int w = threadIdx.x >> 5;
    int wm = w & 1, wn = w >> 1;
    const bf16_t* gA = xb + (long)(blockIdx.y * 128) * 1024;
    const bf16_t* gB = Wb + (long)(blockIdx.x * 128) * 1024;

    stage_copy(sA[0], sB[0], gA, gB, 0);

    v8f acc[4][2] = {};
    for (int kt = 0; kt < 32; ++kt) {
        int cur = kt & 1;
        if (kt + 1 < 32) {
            stage_copy(sA[cur ^ 1], sB[cur ^ 1], gA, gB, (kt + 1) * 32);
            wait_async<4>();
        } else {
            wait_async<0>();
        }
        __syncthreads();
        v16bf a[4], b[2];
#pragma unroll
        for (int mt = 0; mt < 4; ++mt) a[mt] = loadA_lds(sA[cur], wm * 64 + mt * 16);
#pragma unroll
        for (int nt = 0; nt < 2; ++nt) b[nt] = loadB_lds(sB[cur], wn * 32 + nt * 16);
#pragma unroll
        for (int mt = 0; mt < 4; ++mt)
#pragma unroll
            for (int nt = 0; nt < 2; ++nt)
                acc[mt][nt] = wmma_bf16(a[mt], b[nt], acc[mt][nt]);
        __syncthreads();
    }

    int row_base = blockIdx.y * 128 + wm * 64;
    int col_base = blockIdx.x * 128 + wn * 32;
#pragma unroll
    for (int mt = 0; mt < 4; ++mt) {
#pragma unroll
        for (int nt = 0; nt < 2; ++nt) {
            int n  = col_base + nt * 16 + (lane & 15);
            int m0 = row_base + mt * 16 + 8 * (lane >> 4);
            v8f A = acc[mt][nt];
            if (n < 1024) {
#pragma unroll
                for (int v = 0; v < 8; ++v)
                    Qb[(long)(m0 + v) * 1024 + n] = (bf16_t)elu1(A[v]);
            } else if (n < 2048) {
                int c = n - 1024; int hh = c >> 6, d = c & 63;
                int b = m0 >> 13; int s = m0 & 8191;  // tiles never cross batch
                v8bf pk;
#pragma unroll
                for (int v = 0; v < 8; ++v) pk[v] = (bf16_t)elu1(A[v]);
                *(v8bf*)(KT + ((long)(b * 16 + hh) * 64 + d) * 8192 + s) = pk;
            } else {
                int c = n - 2048; int hh = c >> 6, e = c & 63;
                int b = m0 >> 13; int s = m0 & 8191;
                v8bf pk;
#pragma unroll
                for (int v = 0; v < 8; ++v) pk[v] = (bf16_t)A[v];
                *(v8bf*)(VT + ((long)(b * 16 + hh) * 64 + e) * 8192 + s) = pk;
            }
        }
    }
}

// ---------------------------------------------------------------------------
// Kernel 2: per (b,h): KV = K^T @ V over S=8192 (WMMA), Z = sum_s K
// KT/VT are [64][8192] row-major -> both fragments contiguous. Streaming.
// ---------------------------------------------------------------------------
__global__ __launch_bounds__(256) void kv_z_kernel(
    const bf16_t* __restrict__ KT, const bf16_t* __restrict__ VT,
    bf16_t* __restrict__ KVT, float* __restrict__ Z) {
    int bh = blockIdx.x;
    const bf16_t* Kp = KT + (long)bh * 64 * 8192;
    const bf16_t* Vp = VT + (long)bh * 64 * 8192;
    int lane = threadIdx.x & 31;
    int w = threadIdx.x >> 5;
    int mt  = w & 3;            // d tile
    int nt0 = (w >> 2) * 2;     // e tile pair

    v8f acc0 = {}, acc1 = {};
    for (int s0 = 0; s0 < 8192; s0 += 32) {
        __builtin_prefetch(Kp + (long)(mt * 16) * 8192 + s0 + 512, 0, 1);
        __builtin_prefetch(Vp + (long)(nt0 * 16) * 8192 + s0 + 512, 0, 1);
        v16bf a  = loadA_g(Kp, mt * 16, s0, 8192);
        v16bf b0 = loadB_g(Vp, nt0 * 16,      s0, 8192);
        v16bf b1 = loadB_g(Vp, nt0 * 16 + 16, s0, 8192);
        acc0 = wmma_bf16(a, b0, acc0);
        acc1 = wmma_bf16(a, b1, acc1);
    }

    int d0 = mt * 16 + 8 * (lane >> 4);
    {
        int e = nt0 * 16 + (lane & 15);
        v8bf pk;
#pragma unroll
        for (int v = 0; v < 8; ++v) pk[v] = (bf16_t)acc0[v];
        *(v8bf*)(KVT + (long)bh * 4096 + (long)e * 64 + d0) = pk;
    }
    {
        int e = nt0 * 16 + 16 + (lane & 15);
        v8bf pk;
#pragma unroll
        for (int v = 0; v < 8; ++v) pk[v] = (bf16_t)acc1[v];
        *(v8bf*)(KVT + (long)bh * 4096 + (long)e * 64 + d0) = pk;
    }

    // Z[d] = sum_s K[s,d], vectorized b128 reads
    __shared__ float zp[256];
    int d = threadIdx.x >> 2, part = threadIdx.x & 3;
    const v8bf* kr = (const v8bf*)(Kp + (long)d * 8192 + part * 2048);
    float zs = 0.f;
    for (int i = 0; i < 256; ++i) {
        v8bf q = kr[i];
#pragma unroll
        for (int j = 0; j < 8; ++j) zs += (float)q[j];
    }
    zp[threadIdx.x] = zs;
    __syncthreads();
    if (threadIdx.x < 64) {
        int t = threadIdx.x * 4;
        Z[(long)bh * 64 + threadIdx.x] = zp[t] + zp[t + 1] + zp[t + 2] + zp[t + 3];
    }
}

// ---------------------------------------------------------------------------
// Kernel 3: per (b,h), 128-token tile: num = Q @ KV (WMMA, K-dim 64),
// den = Q.Z + 1e-6, attn = num/den -> bf16
// ---------------------------------------------------------------------------
__global__ __launch_bounds__(256) void attn_kernel(
    const bf16_t* __restrict__ Qb, const bf16_t* __restrict__ KVT,
    const float* __restrict__ Z, bf16_t* __restrict__ attnb) {
    int bh = blockIdx.y; int b = bh >> 4, h = bh & 15;
    long mbase = (long)b * 8192 + (long)blockIdx.x * 128;
    const bf16_t* Qp  = Qb + mbase * 1024 + h * 64;   // ld = 1024
    const bf16_t* KVp = KVT + (long)bh * 4096;        // [e][d], ld = 64
    const float*  Zp  = Z + (long)bh * 64;
    int lane = threadIdx.x & 31, w = threadIdx.x >> 5;

    __shared__ float denS[128];
    if (threadIdx.x < 128) {
        const v8bf* q8 = (const v8bf*)(Qp + (long)threadIdx.x * 1024);
        float s = 0.f;
#pragma unroll
        for (int d8 = 0; d8 < 8; ++d8) {
            v8bf q = q8[d8];
#pragma unroll
            for (int j = 0; j < 8; ++j) s += (float)q[j] * Zp[d8 * 8 + j];
        }
        denS[threadIdx.x] = 1.f / (s + 1e-6f);
    }

    v8f acc[4] = {};
#pragma unroll
    for (int k0 = 0; k0 < 64; k0 += 32) {
        v16bf a = loadA_g(Qp, w * 16, k0, 1024);
#pragma unroll
        for (int nt = 0; nt < 4; ++nt)
            acc[nt] = wmma_bf16(a, loadB_g(KVp, nt * 16, k0, 64), acc[nt]);
    }
    __syncthreads();

    int m0 = w * 16 + 8 * (lane >> 4);
#pragma unroll
    for (int nt = 0; nt < 4; ++nt) {
        int col = h * 64 + nt * 16 + (lane & 15);
#pragma unroll
        for (int v = 0; v < 8; ++v) {
            float r = acc[nt][v] * denS[m0 + v];
            attnb[(mbase + m0 + v) * 1024 + col] = (bf16_t)r;
        }
    }
}

// ---------------------------------------------------------------------------
// Kernel 4: out = attn @ Wo^T (M=32768, N=1024, K=1024), LDS staged, f32 out
// ---------------------------------------------------------------------------
__global__ __launch_bounds__(256) void oproj_kernel(
    const bf16_t* __restrict__ attnb, const bf16_t* __restrict__ Wob,
    float* __restrict__ out) {
    __shared__ bf16_t sA[2][128 * LDS_STRIDE];
    __shared__ bf16_t sB[2][128 * LDS_STRIDE];

    int lane = threadIdx.x & 31;
    int w = threadIdx.x >> 5;
    int wm = w & 1, wn = w >> 1;
    const bf16_t* gA = attnb + (long)(blockIdx.y * 128) * 1024;
    const bf16_t* gB = Wob   + (long)(blockIdx.x * 128) * 1024;

    stage_copy(sA[0], sB[0], gA, gB, 0);

    v8f acc[4][2] = {};
    for (int kt = 0; kt < 32; ++kt) {
        int cur = kt & 1;
        if (kt + 1 < 32) {
            stage_copy(sA[cur ^ 1], sB[cur ^ 1], gA, gB, (kt + 1) * 32);
            wait_async<4>();
        } else {
            wait_async<0>();
        }
        __syncthreads();
        v16bf a[4], b[2];
#pragma unroll
        for (int mt = 0; mt < 4; ++mt) a[mt] = loadA_lds(sA[cur], wm * 64 + mt * 16);
#pragma unroll
        for (int nt = 0; nt < 2; ++nt) b[nt] = loadB_lds(sB[cur], wn * 32 + nt * 16);
#pragma unroll
        for (int mt = 0; mt < 4; ++mt)
#pragma unroll
            for (int nt = 0; nt < 2; ++nt)
                acc[mt][nt] = wmma_bf16(a[mt], b[nt], acc[mt][nt]);
        __syncthreads();
    }

    int row_base = blockIdx.y * 128 + wm * 64;
    int col_base = blockIdx.x * 128 + wn * 32;
#pragma unroll
    for (int mt = 0; mt < 4; ++mt)
#pragma unroll
        for (int nt = 0; nt < 2; ++nt) {
            int n  = col_base + nt * 16 + (lane & 15);
            int m0 = row_base + mt * 16 + 8 * (lane >> 4);
#pragma unroll
            for (int v = 0; v < 8; ++v)
                out[(long)(m0 + v) * 1024 + n] = acc[mt][nt][v];
        }
}

// ---------------------------------------------------------------------------
// Launch
// ---------------------------------------------------------------------------
extern "C" void kernel_launch(void* const* d_in, const int* in_sizes, int n_in,
                              void* d_out, int out_size, void* d_ws, size_t ws_size,
                              hipStream_t stream) {
    const float* x    = (const float*)d_in[0];   // [4,8192,1024]
    const float* Wqkv = (const float*)d_in[1];   // [3072,1024]
    const float* Wo   = (const float*)d_in[2];   // [1024,1024]
    float* out = (float*)d_out;                  // [4,8192,1024]

    char* ws = (char*)d_ws;
    bf16_t* wqkvb = (bf16_t*)(ws);                      //  6.0 MB
    bf16_t* wob   = (bf16_t*)(ws + 6291456);            //  2.0 MB
    bf16_t* xb    = (bf16_t*)(ws + 8388608);            // 64.0 MB (reused as attnb)
    bf16_t* Qb    = (bf16_t*)(ws + 75497472);           // 64.0 MB
    bf16_t* KT    = (bf16_t*)(ws + 142606336);          // 64.0 MB
    bf16_t* VT    = (bf16_t*)(ws + 209715200);          // 64.0 MB
    bf16_t* KVT   = (bf16_t*)(ws + 276824064);          //  0.5 MB
    float*  Z     = (float*) (ws + 277348352);          //  16 KB
    bf16_t* attnb = xb;  // xb dead after qkv_gemm; attnb written after

    cvt_bf16x4_kernel<<<(786432 + 255) / 256, 256, 0, stream>>>(Wqkv, wqkvb, 786432);
    cvt_bf16x4_kernel<<<(262144 + 255) / 256, 256, 0, stream>>>(Wo, wob, 262144);
    cvt_bf16x4_kernel<<<(8388608 + 255) / 256, 256, 0, stream>>>(x, xb, 8388608);
    qkv_gemm_kernel<<<dim3(24, 256), 256, 0, stream>>>(xb, wqkvb, Qb, KT, VT);
    kv_z_kernel<<<64, 256, 0, stream>>>(KT, VT, KVT, Z);
    attn_kernel<<<dim3(64, 64), 256, 0, stream>>>(Qb, KVT, Z, attnb);
    oproj_kernel<<<dim3(8, 256), 256, 0, stream>>>(attnb, wob, out);
}
